// AttentionANEWraperChannelsFirstWithCache_60859686584860
// MI455X (gfx1250) — matrix operand order repaired and checked by reading, not compiled, not run on hardware
//
#include <hip/hip_runtime.h>
#include <hip/hip_bf16.h>

typedef unsigned int   u32;
typedef unsigned short u16;

typedef __bf16 bf16_t;
typedef bf16_t bf16x16 __attribute__((ext_vector_type(16)));
typedef float  f32x8   __attribute__((ext_vector_type(8)));
typedef u32    u32x8   __attribute__((ext_vector_type(8)));

#define HH   16
#define KVH  8
#define DD   128
#define HID  2048
#define QQ   1024
#define SEFF 5120      // CACHE_POS + Q; slots beyond are fully masked
#define CPOS 4096

__device__ __forceinline__ u16 f2bf(float f) {
    u32 u = __builtin_bit_cast(u32, f);
    u32 r = u + 0x7fffu + ((u >> 16) & 1u);
    return (u16)(r >> 16);
}

__device__ __forceinline__ u32 ldu32(const u16* p) { return *(const u32*)p; }

// Async global->LDS copy of 8 bytes (gfx1250). %0 = LDS byte address (VGPR),
// %1 = 64-bit global address (VGPR pair). Tracked by ASYNCcnt.
__device__ __forceinline__ void async_cp_b64(u32 lds_addr, const void* gptr) {
    asm volatile("global_load_async_to_lds_b64 %0, %1, off"
                 :: "v"(lds_addr), "v"(gptr) : "memory");
}
__device__ __forceinline__ void wait_async_le4() {
    asm volatile("s_wait_asynccnt 0x4" ::: "memory");
}
__device__ __forceinline__ void wait_async_0() {
    asm volatile("s_wait_asynccnt 0x0" ::: "memory");
}
__device__ __forceinline__ u32 lds_off(const void* p) {
    return (u32)(unsigned long long)p;   // LDS aperture keeps offset in addr[31:0]
}

// ---------------------------------------------------------------------------
// f32 -> bf16 bulk convert
// ---------------------------------------------------------------------------
__global__ void cvt_bf16_kernel(const float* __restrict__ in, u16* __restrict__ out, int n) {
    int i = blockIdx.x * 256 + threadIdx.x;
    if (i < n) out[i] = f2bf(in[i]);
}

// ---------------------------------------------------------------------------
// KV cache (first 4096 slots) -> bf16; K kept (kv,s,d); V transposed (kv,d,s)
// ---------------------------------------------------------------------------
__global__ void cache_cvt_kernel(const float* __restrict__ kc, const float* __restrict__ vc,
                                 u16* __restrict__ kb, u16* __restrict__ vtb) {
    int i = blockIdx.x * 256 + threadIdx.x;           // over 8*4096*128
    if (i >= KVH * CPOS * DD) return;
    int d  = i & (DD - 1);
    int s  = (i >> 7) & (CPOS - 1);
    int kv = i >> 19;
    int src = (kv * 8192 + s) * DD + d;
    kb [(kv * SEFF + s) * DD + d]  = f2bf(kc[src]);
    vtb[(kv * DD + d) * SEFF + s]  = f2bf(vc[src]);
}

// ---------------------------------------------------------------------------
// RoPE on Q: input (c=h*128+d, t) f32 -> qb (h, t, d) bf16
// ---------------------------------------------------------------------------
__global__ void rope_q_kernel(const float* __restrict__ qf, const float* __restrict__ cost,
                              const float* __restrict__ sint, u16* __restrict__ qb) {
    int i = blockIdx.x * 256 + threadIdx.x;           // over 16*128*1024
    if (i >= HH * DD * QQ) return;
    int t = i & (QQ - 1);
    int d = (i >> 10) & (DD - 1);
    int h = i >> 17;
    int c = h * DD + d;
    float v   = qf[c * QQ + t];
    float rot = (d < 64) ? -qf[(c + 64) * QQ + t] : qf[(c - 64) * QQ + t];
    float r   = v * cost[d * QQ + t] + rot * sint[d * QQ + t];
    qb[(h * QQ + t) * DD + d] = f2bf(r);
}

// ---------------------------------------------------------------------------
// RoPE on K: input (c=kv*128+d, t) f32 -> kb band (kv, CPOS+t, d) bf16
// ---------------------------------------------------------------------------
__global__ void rope_k_kernel(const float* __restrict__ kf, const float* __restrict__ cosx,
                              const float* __restrict__ sinx, u16* __restrict__ kb) {
    int i = blockIdx.x * 256 + threadIdx.x;           // over 8*128*1024
    if (i >= KVH * DD * QQ) return;
    int t  = i & (QQ - 1);
    int d  = (i >> 10) & (DD - 1);
    int kv = i >> 17;
    int c  = kv * DD + d;
    float v   = kf[c * QQ + t];
    float rot = (d < 64) ? -kf[(c + 64) * QQ + t] : kf[(c - 64) * QQ + t];
    float r   = v * cosx[t * DD + d] + rot * sinx[t * DD + d];
    kb[(kv * SEFF + (CPOS + t)) * DD + d] = f2bf(r);
}

// ---------------------------------------------------------------------------
// V band: (c=kv*128+d, t) f32 -> vtb (kv, d, CPOS+t) bf16 (transposed)
// ---------------------------------------------------------------------------
__global__ void copy_v_kernel(const float* __restrict__ vf, u16* __restrict__ vtb) {
    int i = blockIdx.x * 256 + threadIdx.x;           // over 8*128*1024
    if (i >= KVH * DD * QQ) return;
    int t  = i & (QQ - 1);
    int d  = (i >> 10) & (DD - 1);
    int kv = i >> 17;
    vtb[(kv * DD + d) * SEFF + (CPOS + t)] = f2bf(vf[(kv * DD + d) * QQ + t]);
}

// ---------------------------------------------------------------------------
// bf16 GEMM: C(MxN) = A(MxK) * B(KxN) + bias, f32 out.
// Block 256 threads = 8 waves; block tile 128x128; wave tile 32x64.
// Double-buffered: A tiles via global_load_async_to_lds_b64 (ASYNCcnt),
// B tiles via VGPR transpose staging; next tile issued during current compute.
// ---------------------------------------------------------------------------
__global__ __launch_bounds__(256)
void gemm_bf16_kernel(const u16* __restrict__ A, const u16* __restrict__ B,
                      const float* __restrict__ bias, float* __restrict__ C,
                      int M, int K, int N) {
    __shared__ alignas(16) u16 As[2][128 * 32];   // [m][k], k contiguous
    __shared__ alignas(16) u16 Bs[2][128 * 32];   // [n][k], k contiguous

    const int tid  = threadIdx.x;
    const int lane = tid & 31;
    const int lr   = lane & 15;
    const int hi   = lane >> 4;
    const int w    = tid >> 5;
    const int wm   = w >> 1;                   // 0..3
    const int wn   = w & 1;                    // 0..1
    const int m0   = blockIdx.y * 128;
    const int n0   = blockIdx.x * 128;

    // Per-thread staging coordinates (4 groups each of A and B per tile).
    int am[4], ak4[4], bn4[4], bkk[4];
#pragma unroll
    for (int it = 0; it < 4; ++it) {
        int idx = tid + it * 256;              // 0..1023
        am[it]  = idx >> 3;                    // A: 128 rows x 8 groups of 4 bf16
        ak4[it] = (idx & 7) * 4;
        bn4[it] = (idx & 31) * 4;              // B: 32 k-rows x 32 groups of 4 bf16
        bkk[it] = idx >> 5;
    }

    f32x8 acc[2][4];
#pragma unroll
    for (int mt = 0; mt < 2; ++mt)
#pragma unroll
        for (int nt = 0; nt < 4; ++nt) acc[mt][nt] = (f32x8)0.0f;

    const int T = K >> 5;                      // number of 32-wide K tiles

    // ---- prologue: async A tile 0 -> As[0]; B tile 0 -> regs
    uint2 breg[4];
#pragma unroll
    for (int it = 0; it < 4; ++it) {
        async_cp_b64(lds_off(&As[0][am[it] * 32 + ak4[it]]),
                     &A[(m0 + am[it]) * K + 0 + ak4[it]]);
        breg[it] = *(const uint2*)&B[(0 + bkk[it]) * N + n0 + bn4[it]];
    }

    for (int i = 0; i < T; ++i) {
        const int p = i & 1;
        // ---- stage B tile i from regs into Bs[p] (bf16 transpose to [n][k])
#pragma unroll
        for (int it = 0; it < 4; ++it) {
            uint2 v = breg[it];
            int n4 = bn4[it], kk = bkk[it];
            Bs[p][(n4 + 0) * 32 + kk] = (u16)(v.x & 0xffffu);
            Bs[p][(n4 + 1) * 32 + kk] = (u16)(v.x >> 16);
            Bs[p][(n4 + 2) * 32 + kk] = (u16)(v.y & 0xffffu);
            Bs[p][(n4 + 3) * 32 + kk] = (u16)(v.y >> 16);
        }
        // ---- issue next tile (A async into back buffer, B into regs)
        if (i + 1 < T) {
            const int k0n = (i + 1) * 32;
#pragma unroll
            for (int it = 0; it < 4; ++it) {
                async_cp_b64(lds_off(&As[p ^ 1][am[it] * 32 + ak4[it]]),
                             &A[(m0 + am[it]) * K + k0n + ak4[it]]);
                breg[it] = *(const uint2*)&B[(k0n + bkk[it]) * N + n0 + bn4[it]];
            }
            wait_async_le4();                  // tile i's 4 copies complete
        } else {
            wait_async_0();
        }
        __syncthreads();

        // ---- compute on buffer p
        bf16x16 af[2], bfr[4];
#pragma unroll
        for (int mt = 0; mt < 2; ++mt) {
            u32x8 u;
#pragma unroll
            for (int j = 0; j < 8; ++j) {
                int kk = ((j < 4) ? 0 : 16) + hi * 8 + (j & 3) * 2;
                u[j] = ldu32(&As[p][(wm * 32 + mt * 16 + lr) * 32 + kk]);
            }
            af[mt] = __builtin_bit_cast(bf16x16, u);
        }
#pragma unroll
        for (int nt = 0; nt < 4; ++nt) {
            u32x8 u;
#pragma unroll
            for (int j = 0; j < 8; ++j) {
                int kk = ((j < 4) ? 0 : 16) + hi * 8 + (j & 3) * 2;
                u[j] = ldu32(&Bs[p][(wn * 64 + nt * 16 + lr) * 32 + kk]);
            }
            bfr[nt] = __builtin_bit_cast(bf16x16, u);
        }
#pragma unroll
        for (int mt = 0; mt < 2; ++mt)
#pragma unroll
            for (int nt = 0; nt < 4; ++nt)
                acc[mt][nt] = __builtin_amdgcn_wmma_f32_16x16x32_bf16(
                    false, af[mt], false, bfr[nt], (short)0, acc[mt][nt], false, false);
        __syncthreads();
    }

#pragma unroll
    for (int mt = 0; mt < 2; ++mt)
#pragma unroll
        for (int nt = 0; nt < 4; ++nt)
#pragma unroll
            for (int r = 0; r < 8; ++r) {
                int m = m0 + wm * 32 + mt * 16 + r + hi * 8;
                int n = n0 + wn * 64 + nt * 16 + lr;
                float v = acc[mt][nt][r] + (bias ? bias[m] : 0.0f);
                C[m * N + n] = v;
            }
}

// ---------------------------------------------------------------------------
// Flash attention: one wave per (head, 16-query tile). Scores = K x Q^T so the
// softmax (key) axis is the C-fragment M axis: 8 in-register lanes + shfl_xor 16.
// O = V^T x P accumulated in 8 C fragments (D=128 rows x 16 queries).
// ---------------------------------------------------------------------------
__global__ __launch_bounds__(32)
void attn_kernel(const u16* __restrict__ qb, const u16* __restrict__ kb,
                 const u16* __restrict__ vtb, u16* __restrict__ outb) {
    __shared__ alignas(8) u16 Pls[16 * 32];    // [q][s], s contiguous

    const int qt = blockIdx.x;                 // 0..63
    const int h  = blockIdx.y;                 // 0..15
    const int kv = h >> 1;
    const int lane = threadIdx.x & 31;
    const int lr = lane & 15, hi = lane >> 4;
    const int t0 = qt * 16;
    const float SC = 0.08838834764831845f;     // 1/sqrt(128)

    // Preload Q as B fragments: (K=d, N=query), qb is (h,t,d) with d contiguous.
    bf16x16 qfr[4];
    const u16* qbase = qb + (h * QQ + t0) * DD;
#pragma unroll
    for (int kk = 0; kk < 4; ++kk) {
        u32x8 u;
#pragma unroll
        for (int j = 0; j < 8; ++j) {
            int d = kk * 32 + ((j < 4) ? 0 : 16) + hi * 8 + (j & 3) * 2;
            u[j] = ldu32(&qbase[lr * DD + d]);
        }
        qfr[kk] = __builtin_bit_cast(bf16x16, u);
    }

    f32x8 of[8];
#pragma unroll
    for (int f = 0; f < 8; ++f) of[f] = (f32x8)0.0f;
    float m_run = -3.0e38f, l_run = 0.0f;

    const int smax = CPOS + t0 + 16;           // exclusive causal bound
    const int nblk = (smax + 31) >> 5;
    const u16* kbh = kb  + kv * SEFF * DD;
    const u16* vbh = vtb + kv * DD * SEFF;

    for (int b = 0; b < nblk; ++b) {
        const int sb = b * 32;
        if (b + 1 < nblk) {
            __builtin_prefetch(&kbh[(sb + 32) * DD], 0, 1);
            __builtin_prefetch(&vbh[sb + 32], 0, 1);
        }

        // ---- scores: C[sub] = K_tile(16 keys x D) * Q^T(D x 16 queries)
        f32x8 c[2];
#pragma unroll
        for (int sub = 0; sub < 2; ++sub) {
            c[sub] = (f32x8)0.0f;
#pragma unroll
            for (int kk = 0; kk < 4; ++kk) {
                u32x8 u;
#pragma unroll
                for (int j = 0; j < 8; ++j) {
                    int d = kk * 32 + ((j < 4) ? 0 : 16) + hi * 8 + (j & 3) * 2;
                    u[j] = ldu32(&kbh[(sb + sub * 16 + lr) * DD + d]);
                }
                c[sub] = __builtin_amdgcn_wmma_f32_16x16x32_bf16(
                    false, __builtin_bit_cast(bf16x16, u), false, qfr[kk],
                    (short)0, c[sub], false, false);
            }
        }

        // ---- causal mask + online softmax (keys on M axis)
        const int tq = t0 + lr;
        float vals[16];
#pragma unroll
        for (int sub = 0; sub < 2; ++sub)
#pragma unroll
            for (int r = 0; r < 8; ++r) {
                int key = sb + sub * 16 + r + hi * 8;
                float x = c[sub][r] * SC;
                vals[sub * 8 + r] = (key <= CPOS + tq) ? x : -1.0e30f;
            }
        float bm = vals[0];
#pragma unroll
        for (int i = 1; i < 16; ++i) bm = fmaxf(bm, vals[i]);
        bm = fmaxf(bm, __shfl_xor(bm, 16, 32));
        float mn   = fmaxf(m_run, bm);
        float corr = __expf(m_run - mn);
        m_run = mn;
        float rs = 0.0f;
#pragma unroll
        for (int i = 0; i < 16; ++i) {
            float p = __expf(vals[i] - mn);
            rs += p;
            int srel = (i >> 3) * 16 + (i & 7) + hi * 8;
            Pls[lr * 32 + srel] = f2bf(p);
        }
        rs += __shfl_xor(rs, 16, 32);
        l_run = l_run * corr + rs;
#pragma unroll
        for (int f = 0; f < 8; ++f) of[f] = of[f] * corr;
        __syncthreads();

        // ---- O += V^T(16 d x 32 s) * P(32 s x 16 q)
        u32x8 up;
#pragma unroll
        for (int j = 0; j < 8; ++j) up[j] = ldu32(&Pls[lr * 32 + ((j < 4) ? 0 : 16) + hi * 8 + (j & 3) * 2]);
        bf16x16 pb = __builtin_bit_cast(bf16x16, up);
#pragma unroll
        for (int f = 0; f < 8; ++f) {
            u32x8 uv;
#pragma unroll
            for (int j = 0; j < 8; ++j) {
                int srel = ((j < 4) ? 0 : 16) + hi * 8 + (j & 3) * 2;
                uv[j] = ldu32(&vbh[(f * 16 + lr) * SEFF + sb + srel]);
            }
            of[f] = __builtin_amdgcn_wmma_f32_16x16x32_bf16(
                false, __builtin_bit_cast(bf16x16, uv), false, pb,
                (short)0, of[f], false, false);
        }
        __syncthreads();
    }

    const float inv = 1.0f / l_run;
#pragma unroll
    for (int f = 0; f < 8; ++f)
#pragma unroll
        for (int r = 0; r < 8; ++r) {
            int d = f * 16 + r + hi * 8;
            outb[(h * DD + d) * QQ + t0 + lr] = f2bf(of[f][r] * inv);
        }
}

// ---------------------------------------------------------------------------
extern "C" void kernel_launch(void* const* d_in, const int* in_sizes, int n_in,
                              void* d_out, int out_size, void* d_ws, size_t ws_size,
                              hipStream_t stream) {
    (void)in_sizes; (void)n_in; (void)out_size; (void)ws_size;
    const float* hidden = (const float*)d_in[0];
    const float* cosx   = (const float*)d_in[1];   // (Q, D)
    const float* sinx   = (const float*)d_in[2];
    const float* cost   = (const float*)d_in[3];   // (D, Q)
    const float* sint   = (const float*)d_in[4];
    // d_in[5] attention_mask: causal mask applied analytically
    const float* kcache = (const float*)d_in[6];
    const float* vcache = (const float*)d_in[7];
    const float* Wq = (const float*)d_in[8];
    const float* bq = (const float*)d_in[9];
    const float* Wk = (const float*)d_in[10];
    const float* bk = (const float*)d_in[11];
    const float* Wv = (const float*)d_in[12];
    const float* bv = (const float*)d_in[13];
    const float* Wo = (const float*)d_in[14];
    float* out = (float*)d_out;

    char* ws = (char*)d_ws;
    u16*   hidb = (u16*)(ws + 0);           // 2048*1024 bf16
    u16*   wqb  = (u16*)(ws + 4194304);     // 2048*2048
    u16*   wkb  = (u16*)(ws + 12582912);    // 1024*2048
    u16*   wvb  = (u16*)(ws + 16777216);    // 1024*2048
    u16*   wob  = (u16*)(ws + 20971520);    // 2048*2048
    float* qf   = (float*)(ws + 29360128);  // 2048*1024 f32
    float* kf   = (float*)(ws + 37748736);  // 1024*1024 f32
    float* vf   = (float*)(ws + 41943040);  // 1024*1024 f32
    u16*   qbp  = (u16*)(ws + 46137344);    // (H,Q,D) bf16
    u16*   kbp  = (u16*)(ws + 50331648);    // (KV,SEFF,D) bf16
    u16*   vtb  = (u16*)(ws + 60817408);    // (KV,D,SEFF) bf16
    u16*   outb = (u16*)(ws + 71303168);    // (2048,1024) bf16

    cvt_bf16_kernel<<<2097152 / 256, 256, 0, stream>>>(hidden, hidb, 2097152);
    cvt_bf16_kernel<<<4194304 / 256, 256, 0, stream>>>(Wq, wqb, 4194304);
    cvt_bf16_kernel<<<2097152 / 256, 256, 0, stream>>>(Wk, wkb, 2097152);
    cvt_bf16_kernel<<<2097152 / 256, 256, 0, stream>>>(Wv, wvb, 2097152);
    cvt_bf16_kernel<<<4194304 / 256, 256, 0, stream>>>(Wo, wob, 4194304);

    gemm_bf16_kernel<<<dim3(8, 16), 256, 0, stream>>>(wqb, hidb, bq, qf, 2048, 2048, 1024);
    gemm_bf16_kernel<<<dim3(8, 8),  256, 0, stream>>>(wkb, hidb, bk, kf, 1024, 2048, 1024);
    gemm_bf16_kernel<<<dim3(8, 8),  256, 0, stream>>>(wvb, hidb, bv, vf, 1024, 2048, 1024);

    cache_cvt_kernel<<<(KVH * CPOS * DD) / 256, 256, 0, stream>>>(kcache, vcache, kbp, vtb);
    rope_q_kernel<<<(HH * DD * QQ) / 256, 256, 0, stream>>>(qf, cost, sint, qbp);
    rope_k_kernel<<<(KVH * DD * QQ) / 256, 256, 0, stream>>>(kf, cosx, sinx, kbp);
    copy_v_kernel<<<(KVH * DD * QQ) / 256, 256, 0, stream>>>(vf, vtb);

    attn_kernel<<<dim3(64, 16), 32, 0, stream>>>(qbp, kbp, vtb, outb);

    gemm_bf16_kernel<<<dim3(8, 16), 256, 0, stream>>>(wob, outb, nullptr, out, 2048, 2048, 1024);
}
